// RingAttention_85607288144445
// MI455X (gfx1250) — matrix operand order repaired and checked by reading, not compiled, not run on hardware
//
#include <hip/hip_runtime.h>

// ---------------------------------------------------------------------------
// Types / helpers
// ---------------------------------------------------------------------------
typedef __attribute__((ext_vector_type(16))) __bf16 v16bf;
typedef __attribute__((ext_vector_type(8)))  float  v8f;

union FragBf {
  v16bf v;
  uint4 q[2];
  unsigned short us[16];
};

__device__ __forceinline__ unsigned short f2bf(float f) {
  unsigned int u = __float_as_uint(f);
  u += 0x7FFFu + ((u >> 16) & 1u);           // round-to-nearest-even
  return (unsigned short)(u >> 16);
}

__device__ __forceinline__ v8f v8f_zero() {
  v8f z;
#pragma unroll
  for (int i = 0; i < 8; ++i) z[i] = 0.0f;
  return z;
}

// CDNA5 async global->LDS copy (ASYNCcnt-tracked, no VGPR landing zone).
// dst-first syntax: global_load_async_to_lds_b128 vdst(lds addr), v[addr:addr+1], off
__device__ __forceinline__ void async_b128(unsigned lds_off, const void* g) {
  asm volatile("global_load_async_to_lds_b128 %0, %1, off"
               :: "v"(lds_off), "v"((unsigned long long)(size_t)g)
               : "memory");
}
__device__ __forceinline__ void wait_async0() {
  asm volatile("s_wait_asynccnt 0x0" ::: "memory");
}
// generic shared ptr -> LDS byte offset (low 32 bits of generic address)
__device__ __forceinline__ unsigned lds_off(const void* p) {
  return (unsigned)(size_t)p;
}

#define WMMA_BF16(A, B, C) \
  __builtin_amdgcn_wmma_f32_16x16x32_bf16(false, (A), false, (B), (short)0, (C), false, false)

#define SCALE_QK 0.08838834764831845f   // 128^-0.5

// ---------------------------------------------------------------------------
// fp32 -> bf16 conversion (x4 vectorized)
// ---------------------------------------------------------------------------
__global__ void cvt_bf16_v4(const float4* __restrict__ src,
                            ushort4* __restrict__ dst, int n4) {
  int i = blockIdx.x * blockDim.x + threadIdx.x;
  if (i < n4) {
    float4 f = src[i];
    ushort4 o;
    o.x = f2bf(f.x); o.y = f2bf(f.y); o.z = f2bf(f.z); o.w = f2bf(f.w);
    dst[i] = o;
  }
}

// ---------------------------------------------------------------------------
// QKV projection GEMM: [4096 x 2048] x [2048 x 6144] -> scatter into
//   q  [2][16][2048][128]  (scaled by 1/sqrt(128))
//   k  [2][16][2048][128]
//   vt [2][16][128][2048]  (transposed for P@V B-fragments)
// Block tile 128x128, 8 waves (4 along M, 2 along N), wave tile 32x64.
// A tile: async global->LDS.  B tile: VGPR path (needs transpose scatter).
// ---------------------------------------------------------------------------
__global__ __launch_bounds__(256) void qkv_gemm(
    const unsigned short* __restrict__ xb,   // [4096][2048] bf16
    const unsigned short* __restrict__ wb,   // [2048][6144] bf16
    unsigned short* __restrict__ qout,
    unsigned short* __restrict__ kout,
    unsigned short* __restrict__ vtout) {
  __shared__ __attribute__((aligned(16))) unsigned short sA[128 * 40];
  __shared__ __attribute__((aligned(16))) unsigned short sB[128 * 40]; // Bt [n][k]

  const int tid  = threadIdx.x;
  const int lane = tid & 31;
  const int wave = tid >> 5;
  const int wm   = wave & 3;      // 0..3 along M
  const int wn   = wave >> 2;     // 0..1 along N
  const int m0   = blockIdx.y * 128;
  const int n0   = blockIdx.x * 128;
  const int half = lane >> 4;
  const int ln   = lane & 15;

  v8f acc[2][4];
#pragma unroll
  for (int i = 0; i < 2; ++i)
#pragma unroll
    for (int j = 0; j < 4; ++j) acc[i][j] = v8f_zero();

  const int ar = tid >> 1, aseg = (tid & 1) * 16;        // A: row, 16-col segment
  const int bk = tid >> 3, bnseg = (tid & 7) * 16;       // B: k row, 16-n segment
  const unsigned sA_off = lds_off(&sA[ar * 40 + aseg]);

  for (int kt = 0; kt < 64; ++kt) {
    const int k0 = kt * 32;
    __syncthreads();
    {  // A tile 128x32 row-major: async global->LDS (2 x b128 per thread)
      const unsigned short* g = xb + (m0 + ar) * 2048 + k0 + aseg;
      async_b128(sA_off,      g);
      async_b128(sA_off + 16, g + 8);
    }
    {  // B tile 32x128 -> LDS transposed [n][k]
      const unsigned short* g = wb + (k0 + bk) * 6144 + n0 + bnseg;
      union { uint4 q[2]; unsigned short us[16]; } t;
      t.q[0] = *(const uint4*)g;
      t.q[1] = *(const uint4*)(g + 8);
#pragma unroll
      for (int i = 0; i < 16; ++i) sB[(bnseg + i) * 40 + bk] = t.us[i];
    }
    if (kt + 1 < 64) {  // prefetch next k-step's tiles into L2/WGP$
      __builtin_prefetch(xb + (m0 + ar) * 2048 + k0 + 32 + aseg, 0, 3);
      __builtin_prefetch(wb + (k0 + 32 + bk) * 6144 + n0 + bnseg, 0, 3);
    }
    wait_async0();
    __syncthreads();

    FragBf a[2];
#pragma unroll
    for (int sm = 0; sm < 2; ++sm) {
      const unsigned short* p = &sA[(wm * 32 + sm * 16 + ln) * 40];
      a[sm].q[0] = *(const uint4*)(p + 16 * half);
      a[sm].q[1] = *(const uint4*)(p + 16 * half + 8);
    }
#pragma unroll
    for (int sn = 0; sn < 4; ++sn) {
      FragBf b;
      const unsigned short* p = &sB[(wn * 64 + sn * 16 + ln) * 40 + 16 * half];
      b.q[0] = *(const uint4*)p;
      b.q[1] = *(const uint4*)(p + 8);
#pragma unroll
      for (int sm = 0; sm < 2; ++sm)
        acc[sm][sn] = WMMA_BF16(a[sm].v, b.v, acc[sm][sn]);
    }
  }

  // epilogue: whole block lies in one (qkv, h) 128-wide column group
  const int qkv = n0 >> 11;            // 0,1,2
  const int h   = (n0 & 2047) >> 7;    // 0..15
#pragma unroll
  for (int sm = 0; sm < 2; ++sm)
#pragma unroll
    for (int sn = 0; sn < 4; ++sn)
#pragma unroll
      for (int r = 0; r < 8; ++r) {
        const int grow = m0 + wm * 32 + sm * 16 + r + 8 * half;  // 0..4095
        const int d    = wn * 64 + sn * 16 + ln;                 // 0..127
        const int bb   = grow >> 11;
        const int nseq = grow & 2047;
        const float vv = acc[sm][sn][r];
        if (qkv == 0)
          qout[(((bb * 16 + h) * 2048) + nseq) * 128 + d] = f2bf(vv * SCALE_QK);
        else if (qkv == 1)
          kout[(((bb * 16 + h) * 2048) + nseq) * 128 + d] = f2bf(vv);
        else
          vtout[(((bb * 16 + h) * 128) + d) * 2048 + nseq] = f2bf(vv);
      }
}

// ---------------------------------------------------------------------------
// Causal flash attention.
// grid = (row_tile 0..31, bh 0..31); 128 threads = 4 waves; wave owns 16 Q rows.
// Q tile 64x128 in registers; K/V tiles streamed via async global->LDS.
// ---------------------------------------------------------------------------
__global__ __launch_bounds__(128) void flash_attn(
    const unsigned short* __restrict__ q,    // [32][2048][128]
    const unsigned short* __restrict__ k,    // [32][2048][128]
    const unsigned short* __restrict__ vt,   // [32][128][2048]
    unsigned short* __restrict__ out) {      // [2][2048][2048] bf16
  __shared__ __attribute__((aligned(16))) unsigned short sK[64 * 136];   // [key][d]
  __shared__ __attribute__((aligned(16))) unsigned short sV[128 * 72];   // [d][key]
  __shared__ __attribute__((aligned(16))) unsigned short sP[4][16 * 72]; // per-wave P

  const int bh   = blockIdx.y;
  const int rt   = blockIdx.x;
  const int tid  = threadIdx.x;
  const int lane = tid & 31;
  const int wave = tid >> 5;
  const int half = lane >> 4;
  const int ln   = lane & 15;
  const int rowBase = rt * 64;

  // Q A-fragments straight from global (row-major matches A layout)
  FragBf aQ[4];
  {
    const unsigned short* qp = q + (bh * 2048 + rowBase + wave * 16 + ln) * 128;
#pragma unroll
    for (int kc = 0; kc < 4; ++kc) {
      aQ[kc].q[0] = *(const uint4*)(qp + kc * 32 + 16 * half);
      aQ[kc].q[1] = *(const uint4*)(qp + kc * 32 + 16 * half + 8);
    }
  }

  v8f o[8];
#pragma unroll
  for (int dd = 0; dd < 8; ++dd) o[dd] = v8f_zero();
  float rmax[8], rsum[8];
#pragma unroll
  for (int r = 0; r < 8; ++r) { rmax[r] = -3.0e38f; rsum[r] = 0.0f; }

  // per-thread tile-load mapping
  const int krow = tid >> 1, kseg = (tid & 1) * 64;      // K: row, 64-elem half
  const unsigned sK_off = lds_off(&sK[krow * 136 + kseg]);
  const unsigned sV_off = lds_off(&sV[tid * 72]);

  for (int j = 0; j <= rt; ++j) {
    const int kBase = j * 64;
    __syncthreads();
    {  // K tile: 64 keys x 128 d, row-major (async)
      const unsigned short* g = k + (bh * 2048 + kBase + krow) * 128 + kseg;
#pragma unroll
      for (int i = 0; i < 8; ++i) async_b128(sK_off + i * 16, g + i * 8);
    }
    {  // V tile (pre-transposed in global): 128 d x 64 keys (async)
      const unsigned short* g = vt + (bh * 128 + tid) * 2048 + kBase;
#pragma unroll
      for (int i = 0; i < 8; ++i) async_b128(sV_off + i * 16, g + i * 8);
    }
    wait_async0();
    __syncthreads();

    // S = Q @ K^T : 16 x 64 per wave
    v8f s[4];
#pragma unroll
    for (int sn = 0; sn < 4; ++sn) {
      s[sn] = v8f_zero();
#pragma unroll
      for (int kc = 0; kc < 4; ++kc) {
        FragBf b;
        const unsigned short* p = &sK[(sn * 16 + ln) * 136 + kc * 32 + 16 * half];
        b.q[0] = *(const uint4*)p;
        b.q[1] = *(const uint4*)(p + 8);
        s[sn] = WMMA_BF16(aQ[kc].v, b.v, s[sn]);
      }
    }

    // causal mask on the diagonal block
    if (j == rt) {
#pragma unroll
      for (int sn = 0; sn < 4; ++sn) {
        const int key = kBase + sn * 16 + ln;
#pragma unroll
        for (int r = 0; r < 8; ++r) {
          const int row = rowBase + wave * 16 + r + 8 * half;
          if (key > row) s[sn][r] = -3.0e38f;
        }
      }
    }

    // online softmax (rows live in element r, lanes of a 16-lane half)
    float alpha[8];
#pragma unroll
    for (int r = 0; r < 8; ++r) {
      float m = fmaxf(fmaxf(s[0][r], s[1][r]), fmaxf(s[2][r], s[3][r]));
      m = fmaxf(m, __shfl_xor(m, 1, 32));
      m = fmaxf(m, __shfl_xor(m, 2, 32));
      m = fmaxf(m, __shfl_xor(m, 4, 32));
      m = fmaxf(m, __shfl_xor(m, 8, 32));
      const float nm = fmaxf(rmax[r], m);
      alpha[r] = __expf(rmax[r] - nm);
      rmax[r] = nm;
      float rs = 0.0f;
#pragma unroll
      for (int sn = 0; sn < 4; ++sn) {
        const float pv = __expf(s[sn][r] - nm);
        s[sn][r] = pv;
        rs += pv;
      }
      rs += __shfl_xor(rs, 1, 32);
      rs += __shfl_xor(rs, 2, 32);
      rs += __shfl_xor(rs, 4, 32);
      rs += __shfl_xor(rs, 8, 32);
      rsum[r] = rsum[r] * alpha[r] + rs;
    }
#pragma unroll
    for (int dd = 0; dd < 8; ++dd)
#pragma unroll
      for (int r = 0; r < 8; ++r) o[dd][r] *= alpha[r];

    // re-layout P (C-frag) -> A-frag through per-wave LDS
    unsigned short* pP = &sP[wave][0];
#pragma unroll
    for (int sn = 0; sn < 4; ++sn)
#pragma unroll
      for (int r = 0; r < 8; ++r)
        pP[(r + 8 * half) * 72 + sn * 16 + ln] = f2bf(s[sn][r]);

    // O += P @ V
#pragma unroll
    for (int kc = 0; kc < 2; ++kc) {
      FragBf aP;
      const unsigned short* pr = &sP[wave][ln * 72 + kc * 32];
      aP.q[0] = *(const uint4*)(pr + 16 * half);
      aP.q[1] = *(const uint4*)(pr + 16 * half + 8);
#pragma unroll
      for (int dd = 0; dd < 8; ++dd) {
        FragBf bV;
        const unsigned short* pv = &sV[(dd * 16 + ln) * 72 + kc * 32 + 16 * half];
        bV.q[0] = *(const uint4*)pv;
        bV.q[1] = *(const uint4*)(pv + 8);
        o[dd] = WMMA_BF16(aP.v, bV.v, o[dd]);
      }
    }
  }

  // normalize + store bf16 into [b][n][h*128+d]
  const int b = bh >> 4, h = bh & 15;
#pragma unroll
  for (int dd = 0; dd < 8; ++dd)
#pragma unroll
    for (int r = 0; r < 8; ++r) {
      const int row = rowBase + wave * 16 + r + 8 * half;
      const int d   = dd * 16 + ln;
      out[(b * 2048 + row) * 2048 + h * 128 + d] = f2bf(o[dd][r] / rsum[r]);
    }
}

// ---------------------------------------------------------------------------
// Output projection GEMM: [4096 x 2048] x [2048 x 2048] -> fp32 d_out
// ---------------------------------------------------------------------------
__global__ __launch_bounds__(256) void out_gemm(
    const unsigned short* __restrict__ ab,   // [4096][2048] bf16
    const unsigned short* __restrict__ wb,   // [2048][2048] bf16
    float* __restrict__ dout) {
  __shared__ __attribute__((aligned(16))) unsigned short sA[128 * 40];
  __shared__ __attribute__((aligned(16))) unsigned short sB[128 * 40];

  const int tid  = threadIdx.x;
  const int lane = tid & 31;
  const int wave = tid >> 5;
  const int wm   = wave & 3;
  const int wn   = wave >> 2;
  const int m0   = blockIdx.y * 128;
  const int n0   = blockIdx.x * 128;
  const int half = lane >> 4;
  const int ln   = lane & 15;

  v8f acc[2][4];
#pragma unroll
  for (int i = 0; i < 2; ++i)
#pragma unroll
    for (int j = 0; j < 4; ++j) acc[i][j] = v8f_zero();

  const int ar = tid >> 1, aseg = (tid & 1) * 16;
  const int bk = tid >> 3, bnseg = (tid & 7) * 16;
  const unsigned sA_off = lds_off(&sA[ar * 40 + aseg]);

  for (int kt = 0; kt < 64; ++kt) {
    const int k0 = kt * 32;
    __syncthreads();
    {  // A tile: async global->LDS
      const unsigned short* g = ab + (m0 + ar) * 2048 + k0 + aseg;
      async_b128(sA_off,      g);
      async_b128(sA_off + 16, g + 8);
    }
    {  // B tile -> LDS transposed
      const unsigned short* g = wb + (k0 + bk) * 2048 + n0 + bnseg;
      union { uint4 q[2]; unsigned short us[16]; } t;
      t.q[0] = *(const uint4*)g;
      t.q[1] = *(const uint4*)(g + 8);
#pragma unroll
      for (int i = 0; i < 16; ++i) sB[(bnseg + i) * 40 + bk] = t.us[i];
    }
    if (kt + 1 < 64) {
      __builtin_prefetch(ab + (m0 + ar) * 2048 + k0 + 32 + aseg, 0, 3);
      __builtin_prefetch(wb + (k0 + 32 + bk) * 2048 + n0 + bnseg, 0, 3);
    }
    wait_async0();
    __syncthreads();

    FragBf a[2];
#pragma unroll
    for (int sm = 0; sm < 2; ++sm) {
      const unsigned short* p = &sA[(wm * 32 + sm * 16 + ln) * 40];
      a[sm].q[0] = *(const uint4*)(p + 16 * half);
      a[sm].q[1] = *(const uint4*)(p + 16 * half + 8);
    }
#pragma unroll
    for (int sn = 0; sn < 4; ++sn) {
      FragBf b;
      const unsigned short* p = &sB[(wn * 64 + sn * 16 + ln) * 40 + 16 * half];
      b.q[0] = *(const uint4*)p;
      b.q[1] = *(const uint4*)(p + 8);
#pragma unroll
      for (int sm = 0; sm < 2; ++sm)
        acc[sm][sn] = WMMA_BF16(a[sm].v, b.v, acc[sm][sn]);
    }
  }

#pragma unroll
  for (int sm = 0; sm < 2; ++sm)
#pragma unroll
    for (int sn = 0; sn < 4; ++sn)
#pragma unroll
      for (int r = 0; r < 8; ++r) {
        const int grow = m0 + wm * 32 + sm * 16 + r + 8 * half;
        const int gcol = n0 + wn * 64 + sn * 16 + ln;
        dout[grow * 2048 + gcol] = acc[sm][sn][r];
      }
}

// ---------------------------------------------------------------------------
// Launch
// ---------------------------------------------------------------------------
extern "C" void kernel_launch(void* const* d_in, const int* in_sizes, int n_in,
                              void* d_out, int out_size, void* d_ws, size_t ws_size,
                              hipStream_t stream) {
  (void)in_sizes; (void)n_in; (void)out_size; (void)ws_size;
  const float* x     = (const float*)d_in[0];   // [2][2048][2048]
  const float* w_qkv = (const float*)d_in[1];   // [2048][6144]
  const float* w_out = (const float*)d_in[2];   // [2048][2048]
  float* out = (float*)d_out;

  char* ws = (char*)d_ws;
  unsigned short* xb    = (unsigned short*)(ws + 0);          //  16 MB
  unsigned short* wqkvb = (unsigned short*)(ws + 16777216);   //  24 MB
  unsigned short* woutb = (unsigned short*)(ws + 41943040);   //   8 MB
  unsigned short* qb    = (unsigned short*)(ws + 50331648);   //  16 MB
  unsigned short* kb    = (unsigned short*)(ws + 67108864);   //  16 MB
  unsigned short* vtb   = (unsigned short*)(ws + 83886080);   //  16 MB
  unsigned short* aob   = (unsigned short*)(ws + 100663296);  //  16 MB

  cvt_bf16_v4<<<(8388608 / 4 + 255) / 256, 256, 0, stream>>>(
      (const float4*)x, (ushort4*)xb, 8388608 / 4);
  cvt_bf16_v4<<<(12582912 / 4 + 255) / 256, 256, 0, stream>>>(
      (const float4*)w_qkv, (ushort4*)wqkvb, 12582912 / 4);
  cvt_bf16_v4<<<(4194304 / 4 + 255) / 256, 256, 0, stream>>>(
      (const float4*)w_out, (ushort4*)woutb, 4194304 / 4);

  qkv_gemm<<<dim3(48, 32), 256, 0, stream>>>(xb, wqkvb, qb, kb, vtb);
  flash_attn<<<dim3(32, 32), 128, 0, stream>>>(qb, kb, vtb, aob);
  out_gemm<<<dim3(16, 32), 256, 0, stream>>>(aob, woutb, out);
}
